// DeformableSpatialAttention_9156870275468
// MI455X (gfx1250) — matrix operand order repaired and checked by reading, not compiled, not run on hardware
//
#include <hip/hip_runtime.h>
#include <hip/hip_bf16.h>

// ---------------- problem constants ----------------
constexpr int B0n = 4, NFn = 3, En = 256, NHn = 8, NPn = 12, Ln = 8;
constexpr int HH = 64, WW = 64;
constexpr int DPH = En / NHn;          // 32
constexpr int Cc = DPH / NHn;          // 4
constexpr int NQn = HH * WW;           // 4096
constexpr int BSn = B0n * NFn;         // 12
constexpr int Mrows = BSn * NQn;       // 49152

typedef __attribute__((ext_vector_type(16))) _Float16 v16h;
typedef __attribute__((ext_vector_type(8)))  _Float16 v8h;
typedef __attribute__((ext_vector_type(8)))  float    v8f;

// ---------------- layout transform: (B0,E,NF,H,W) -> (BS,NQ,E) ----------------
__global__ void to_seq_kernel(const float* __restrict__ fea, float* __restrict__ X) {
  size_t i = (size_t)blockIdx.x * blockDim.x + threadIdx.x;
  const size_t total = (size_t)Mrows * En;
  if (i >= total) return;
  int e = (int)(i % En);
  size_t r = i / En;
  int q = (int)(r % NQn);
  int b = (int)(r / NQn);
  int b0 = b / NFn, f = b % NFn;
  X[i] = fea[(((size_t)b0 * En + e) * NFn + f) * NQn + q];
}

// ---------------- positional embedding: (NQ,E) = cat(col[x], row[y]) ----------------
__global__ void pos_kernel(const float* __restrict__ row, const float* __restrict__ col,
                           float* __restrict__ qpos) {
  int i = blockIdx.x * blockDim.x + threadIdx.x;
  if (i >= NQn * En) return;
  int e = i % En, q = i / En;
  int y = q / WW, x = q % WW;
  qpos[i] = (e < En / 2) ? col[x * (En / 2) + e] : row[y * (En / 2) + (e - En / 2)];
}

// ---------------- weight prep: (L,K,N) f32 -> (L,N,K) f16 ----------------
__global__ void wprep_kernel(const float* __restrict__ Wsrc, _Float16* __restrict__ Dst,
                             int K, int N) {
  size_t i = (size_t)blockIdx.x * blockDim.x + threadIdx.x;
  size_t total = (size_t)Ln * K * N;
  if (i >= total) return;
  int n = (int)(i % N);
  size_t r = i / N;
  int k = (int)(r % K);
  int l = (int)(r / K);
  Dst[((size_t)l * N + n) * K + k] = (_Float16)Wsrc[i];
}

// ---------------- A prep: f16(X) or f16(X + qpos broadcast over batch) ----------------
template <bool HASQ>
__global__ void cvt16_kernel(const float* __restrict__ X, const float* __restrict__ qpos,
                             _Float16* __restrict__ Y) {
  size_t i = (size_t)blockIdx.x * blockDim.x + threadIdx.x;
  const size_t total = (size_t)Mrows * En;
  if (i >= total) return;
  float v = X[i];
  if (HASQ) v += qpos[i & ((size_t)NQn * En - 1)];  // NQ*E is a power of two
  Y[i] = (_Float16)v;
}

// ---------------- WMMA GEMM with LDS-staged B panel ----------------
// Block = 8 waves = 8 consecutive M-tiles sharing one (NT x 16)-column B strip.
// B strip is cooperatively staged into LDS once, then each wave runs the
// pipelined K-loop: A fragments from global (f16), B fragments from LDS.
// All NT B fragments of a K-step are loaded into distinct registers before the
// WMMA burst so DS waits stagger (dscnt N..0) instead of stalling per WMMA.
// Grid covers tiles exactly (no early exit) -> EXEC all-ones for WMMA.
template <int NT, bool HASRES>
__global__ void gemm_wmma_kernel(const _Float16* __restrict__ A,   // (M,K) f16
                                 const _Float16* __restrict__ Bt,  // (N,K) f16
                                 const float* __restrict__ bias,   // (N)
                                 const float* __restrict__ residual, // (M,N) or unused
                                 float* __restrict__ C,
                                 int N, int K) {
  extern __shared__ _Float16 Bs[];              // NT*16*K halves
  const int lane = threadIdx.x & 31;
  const int wave = threadIdx.x >> 5;
  const int tilesN = N >> 4;
  const int groupsN = tilesN / NT;
  const int bm = blockIdx.x / groupsN;          // M-block (8 tiles)
  const int gn = blockIdx.x - bm * groupsN;     // N-strip group
  const int lo = lane & 15;
  const int hi = lane >> 4;

  // ---- stage B strip: NT*16 rows x K halves, vectorized 16B copies ----
  {
    const int S8 = (NT * 16 * K) / 8;           // # of 8-half (16B) chunks
    const uint4* src = (const uint4*)(Bt + (size_t)(gn * NT * 16) * K);
    uint4* dst = (uint4*)Bs;
    for (int i = threadIdx.x; i < S8; i += (int)blockDim.x) dst[i] = src[i];
  }
  __syncthreads();

  const int tm = bm * 8 + wave;                 // this wave's M-tile
  // A fragment, contiguous K order (matches B):
  //  lanes 0-15 : halves 0-15 -> K=k0+0..15
  //  lanes16-31 : halves 0-15 -> K=k0+16..31
  const _Float16* aptr = A + (size_t)(tm * 16 + lo) * K + hi * 16;
  // B fragment: lane = column, 16 contiguous K starting at hi*16 (32B LDS load)
  const _Float16* bbase = Bs + (size_t)lo * K + hi * 16;

  v8f acc[NT] = {};
  for (int k0 = 0; k0 < K; k0 += 32) {
    v8h a0 = *(const v8h*)(aptr + k0);
    v8h a1 = *(const v8h*)(aptr + k0 + 8);
    v16h a = __builtin_shufflevector(a0, a1, 0, 1, 2, 3, 4, 5, 6, 7,
                                     8, 9, 10, 11, 12, 13, 14, 15);
    // Issue all B-fragment loads first (multi-buffer), then the WMMA burst.
    v16h bf[NT];
#pragma unroll
    for (int j = 0; j < NT; ++j)
      bf[j] = *(const v16h*)(bbase + (size_t)j * 16 * K + k0);
#pragma unroll
    for (int j = 0; j < NT; ++j)
      acc[j] = __builtin_amdgcn_wmma_f32_16x16x32_f16(false, a, false, bf[j],
                                                      (short)0, acc[j], false, false);
  }
#pragma unroll
  for (int j = 0; j < NT; ++j) {
    const int n = (gn * NT + j) * 16 + lo;
    const float bn = bias[n];
#pragma unroll
    for (int i = 0; i < 8; ++i) {
      int m = tm * 16 + hi * 8 + i;
      size_t idx = (size_t)m * N + n;
      float v = acc[j][i] + bn;
      if (HASRES) v += residual[idx];
      C[idx] = v;
    }
  }
}

// ---------------- softmax over NP=12 (in place), att: (BS,NQ,NH,NP) ----------------
__global__ void softmax_np_kernel(float* __restrict__ att) {
  int i = blockIdx.x * blockDim.x + threadIdx.x;
  if (i >= BSn * NQn * NHn) return;
  float* p = att + (size_t)i * NPn;
  float mx = p[0];
#pragma unroll
  for (int j = 1; j < NPn; ++j) mx = fmaxf(mx, p[j]);
  float e[NPn], s = 0.f;
#pragma unroll
  for (int j = 0; j < NPn; ++j) { e[j] = __expf(p[j] - mx); s += e[j]; }
  float inv = 1.f / s;
#pragma unroll
  for (int j = 0; j < NPn; ++j) p[j] = e[j] * inv;
}

// ---------------- deformable bilinear sample + attention-weighted sum ----------------
// v:(BS,NQ,32) off:(BS,NQ,NH,NP,2) att:(BS,NQ,NH,NP) -> outs:(BS,NQ,NH*C) as f16
__global__ void sample_kernel(const float* __restrict__ v, const float* __restrict__ off,
                              const float* __restrict__ att, _Float16* __restrict__ outs) {
  int i = blockIdx.x * blockDim.x + threadIdx.x;
  if (i >= BSn * NQn * NHn) return;
  int h = i % NHn;
  int q = (i / NHn) % NQn;
  int b = i / (NHn * NQn);
  float rx = ((q % WW) + 0.5f) * (1.f / WW);
  float ry = ((q / WW) + 0.5f) * (1.f / HH);
  const float* offp = off + ((size_t)(b * NQn + q) * NHn + h) * (NPn * 2);
  const float* attp = att + ((size_t)(b * NQn + q) * NHn + h) * NPn;
  const float* vb = v + (size_t)b * NQn * DPH + h * Cc;
  float acc0 = 0.f, acc1 = 0.f, acc2 = 0.f, acc3 = 0.f;
  for (int p = 0; p < NPn; ++p) {
    float gx = 2.f * (rx + offp[2 * p]     * (1.f / WW)) - 1.f;
    float gy = 2.f * (ry + offp[2 * p + 1] * (1.f / HH)) - 1.f;
    float x = (gx + 1.f) * (WW * 0.5f) - 0.5f;
    float y = (gy + 1.f) * (HH * 0.5f) - 0.5f;
    float x0f = floorf(x), y0f = floorf(y);
    int x0 = (int)x0f, y0 = (int)y0f;
    float wx1 = x - x0f, wy1 = y - y0f;
    float aw = attp[p];
#pragma unroll
    for (int dy = 0; dy < 2; ++dy) {
#pragma unroll
      for (int dx = 0; dx < 2; ++dx) {
        int xi = x0 + dx, yi = y0 + dy;
        if (xi < 0 || xi >= WW || yi < 0 || yi >= HH) continue;
        float wgt = (dx ? wx1 : 1.f - wx1) * (dy ? wy1 : 1.f - wy1) * aw;
        const float4 vv = *(const float4*)(vb + (size_t)(yi * WW + xi) * DPH);
        acc0 += wgt * vv.x; acc1 += wgt * vv.y; acc2 += wgt * vv.z; acc3 += wgt * vv.w;
      }
    }
  }
  _Float16* op = outs + (size_t)(b * NQn + q) * DPH + h * Cc;
  op[0] = (_Float16)acc0; op[1] = (_Float16)acc1;
  op[2] = (_Float16)acc2; op[3] = (_Float16)acc3;
}

// ---------------- host side ----------------
static inline int cdiv(long long a, long long b) { return (int)((a + b - 1) / b); }

extern "C" void kernel_launch(void* const* d_in, const int* in_sizes, int n_in,
                              void* d_out, int out_size, void* d_ws, size_t ws_size,
                              hipStream_t stream) {
  (void)in_sizes; (void)n_in; (void)out_size; (void)ws_size;
  const float* rgb_fea = (const float*)d_in[0];
  const float* ir_fea  = (const float*)d_in[1];
  const float* rgb_row = (const float*)d_in[2];
  const float* rgb_col = (const float*)d_in[3];
  const float* ir_row  = (const float*)d_in[4];
  const float* ir_col  = (const float*)d_in[5];
  const float* Woff[2] = {(const float*)d_in[6],  (const float*)d_in[14]};
  const float* boff[2] = {(const float*)d_in[7],  (const float*)d_in[15]};
  const float* Watt[2] = {(const float*)d_in[8],  (const float*)d_in[16]};
  const float* batt[2] = {(const float*)d_in[9],  (const float*)d_in[17]};
  const float* Wv[2]   = {(const float*)d_in[10], (const float*)d_in[18]};
  const float* bv[2]   = {(const float*)d_in[11], (const float*)d_in[19]};
  const float* Wo[2]   = {(const float*)d_in[12], (const float*)d_in[20]};
  const float* bo[2]   = {(const float*)d_in[13], (const float*)d_in[21]};

  // ---- workspace carving ----
  char* ws = (char*)d_ws;
  size_t cur = 0;
  auto carve = [&](size_t bytes) {
    cur = (cur + 255) & ~(size_t)255;
    void* p = ws + cur;
    cur += bytes;
    return p;
  };
  const size_t bigBytes = (size_t)Mrows * En * sizeof(float);
  float*    buf1_rgb = (float*)carve(bigBytes);
  float*    buf1_ir  = (float*)carve(bigBytes);
  float*    qpos_rgb = (float*)carve((size_t)NQn * En * sizeof(float));
  float*    qpos_ir  = (float*)carve((size_t)NQn * En * sizeof(float));
  _Float16* a16_q    = (_Float16*)carve((size_t)Mrows * En * sizeof(_Float16));
  _Float16* a16_val  = (_Float16*)carve((size_t)Mrows * En * sizeof(_Float16));
  float*    vbuf     = (float*)carve((size_t)Mrows * DPH * sizeof(float));
  float*    offbuf   = (float*)carve((size_t)Mrows * (NHn * NPn * 2) * sizeof(float));
  float*    attbuf   = (float*)carve((size_t)Mrows * (NHn * NPn) * sizeof(float));
  _Float16* outsbuf  = (_Float16*)carve((size_t)Mrows * DPH * sizeof(_Float16));
  _Float16 *Woff_t[2], *Watt_t[2], *Wv_t[2], *Wo_t[2];
  for (int m = 0; m < 2; ++m) {
    Woff_t[m] = (_Float16*)carve((size_t)Ln * (NHn * NPn * 2) * En * sizeof(_Float16));
    Watt_t[m] = (_Float16*)carve((size_t)Ln * (NHn * NPn) * En * sizeof(_Float16));
    Wv_t[m]   = (_Float16*)carve((size_t)Ln * DPH * En * sizeof(_Float16));
    Wo_t[m]   = (_Float16*)carve((size_t)Ln * En * DPH * sizeof(_Float16));
  }

  float* out_rgb = (float*)d_out;
  float* out_ir  = out_rgb + (size_t)Mrows * En;

  const int TPB = 256;
  const long long ME = (long long)Mrows * En;

  // ---- one-time-per-call prep ----
  to_seq_kernel<<<cdiv(ME, TPB), TPB, 0, stream>>>(rgb_fea, out_rgb);
  to_seq_kernel<<<cdiv(ME, TPB), TPB, 0, stream>>>(ir_fea, out_ir);
  pos_kernel<<<cdiv(NQn * En, TPB), TPB, 0, stream>>>(rgb_row, rgb_col, qpos_rgb);
  pos_kernel<<<cdiv(NQn * En, TPB), TPB, 0, stream>>>(ir_row, ir_col, qpos_ir);
  for (int m = 0; m < 2; ++m) {
    wprep_kernel<<<cdiv((long long)Ln * En * 192, TPB), TPB, 0, stream>>>(Woff[m], Woff_t[m], En, 192);
    wprep_kernel<<<cdiv((long long)Ln * En * 96,  TPB), TPB, 0, stream>>>(Watt[m], Watt_t[m], En, 96);
    wprep_kernel<<<cdiv((long long)Ln * En * 32,  TPB), TPB, 0, stream>>>(Wv[m],   Wv_t[m],   En, 32);
    wprep_kernel<<<cdiv((long long)Ln * 32 * En,  TPB), TPB, 0, stream>>>(Wo[m],   Wo_t[m],   32, En);
  }

  // ---- GEMM dispatch: block = 8 M-tiles x one NT*16-column strip ----
  const int tilesM = Mrows / 16;                // 3072, divisible by 8
  auto gemm = [&](const _Float16* A16, const _Float16* Bt, const float* bias,
                  const float* res, float* Cb, int N, int K) {
    auto blocks = [&](int NT) { return (tilesM / 8) * ((N / 16) / NT); };
    auto smem = [&](int NT) { return (size_t)NT * 16 * K * sizeof(_Float16); };
    if (N == 192) {
      if (res) gemm_wmma_kernel<4, true ><<<blocks(4), TPB, smem(4), stream>>>(A16, Bt, bias, res, Cb, N, K);
      else     gemm_wmma_kernel<4, false><<<blocks(4), TPB, smem(4), stream>>>(A16, Bt, bias, res, Cb, N, K);
    } else if (N == 96) {
      if (res) gemm_wmma_kernel<6, true ><<<blocks(6), TPB, smem(6), stream>>>(A16, Bt, bias, res, Cb, N, K);
      else     gemm_wmma_kernel<6, false><<<blocks(6), TPB, smem(6), stream>>>(A16, Bt, bias, res, Cb, N, K);
    } else if (N == 32) {
      if (res) gemm_wmma_kernel<2, true ><<<blocks(2), TPB, smem(2), stream>>>(A16, Bt, bias, res, Cb, N, K);
      else     gemm_wmma_kernel<2, false><<<blocks(2), TPB, smem(2), stream>>>(A16, Bt, bias, res, Cb, N, K);
    } else {  // N == 256
      if (res) gemm_wmma_kernel<4, true ><<<blocks(4), TPB, smem(4), stream>>>(A16, Bt, bias, res, Cb, N, K);
      else     gemm_wmma_kernel<4, false><<<blocks(4), TPB, smem(4), stream>>>(A16, Bt, bias, res, Cb, N, K);
    }
  };

  // ---- layers: ping-pong between d_out halves (buf0) and ws (buf1) ----
  float* curp[2] = {out_rgb, out_ir};
  float* newp[2] = {buf1_rgb, buf1_ir};
  const float* qposm[2] = {qpos_rgb, qpos_ir};

  for (int l = 0; l < Ln; ++l) {
    for (int m = 0; m < 2; ++m) {
      const float* query = curp[m];
      const float* value = curp[1 - m];
      // A-operand prep: f16(value) and f16(query + qpos)
      cvt16_kernel<false><<<cdiv(ME, TPB), TPB, 0, stream>>>(value, nullptr, a16_val);
      cvt16_kernel<true ><<<cdiv(ME, TPB), TPB, 0, stream>>>(query, qposm[m], a16_q);
      // v = value @ Wv + bv
      gemm(a16_val, Wv_t[m] + (size_t)l * DPH * En, bv[m] + (size_t)l * DPH,
           nullptr, vbuf, DPH, En);
      // off = (query+qpos) @ Woff + boff
      gemm(a16_q, Woff_t[m] + (size_t)l * 192 * En, boff[m] + (size_t)l * 192,
           nullptr, offbuf, 192, En);
      // att = softmax((query+qpos) @ Watt + batt)
      gemm(a16_q, Watt_t[m] + (size_t)l * 96 * En, batt[m] + (size_t)l * 96,
           nullptr, attbuf, 96, En);
      softmax_np_kernel<<<cdiv(BSn * NQn * NHn, TPB), TPB, 0, stream>>>(attbuf);
      // deformable sampling -> f16 A for the output projection
      sample_kernel<<<cdiv(BSn * NQn * NHn, TPB), TPB, 0, stream>>>(vbuf, offbuf, attbuf, outsbuf);
      // new = outs @ Wo + bo + query
      gemm(outsbuf, Wo_t[m] + (size_t)l * En * DPH, bo[m] + (size_t)l * En,
           query, newp[m], En, DPH);
    }
    float* t0 = curp[0]; float* t1 = curp[1];
    curp[0] = newp[0]; curp[1] = newp[1];
    newp[0] = t0; newp[1] = t1;
  }
  // Ln = 8 (even) -> final results land back in d_out halves (rgb, ir).
}
